// SpatialTemporalMaskModelAttention_40389872451952
// MI455X (gfx1250) — matrix operand order repaired and checked by reading, not compiled, not run on hardware
//
#include <hip/hip_runtime.h>

#define HW_ 8192
#define HH_ 64
#define WW_ 128

typedef __attribute__((ext_vector_type(16))) __bf16 v16bf;
typedef __attribute__((ext_vector_type(8)))  float  v8f;
typedef __attribute__((ext_vector_type(4)))  unsigned v4u;
typedef __attribute__((ext_vector_type(4)))  int  v4i;
typedef __attribute__((ext_vector_type(8)))  int  v8i;

#if defined(__has_builtin)
#if __has_builtin(__builtin_amdgcn_tensor_load_to_lds) && __has_builtin(__builtin_amdgcn_s_wait_tensorcnt)
#define HAVE_TDM 1
#endif
#endif
#ifndef HAVE_TDM
#define HAVE_TDM 0
#endif

__device__ __forceinline__ __bf16 f2bf(float f) {
  unsigned u = __builtin_bit_cast(unsigned, f);
  u = (u + 0x7FFFu + ((u >> 16) & 1u)) >> 16;   // round-to-nearest-even
  unsigned short s = (unsigned short)u;
  return __builtin_bit_cast(__bf16, s);
}

#if HAVE_TDM
// Issue one TDM 2D copy: nrows rows of rowUnits 8-byte elements, row stride
// strideUnits (8B units), global -> LDS at byte offset ldsByte (contiguous).
// D# layout per CDNA5 ISA 8.3/8.4 (group0: count/lds/global/type; group1:
// data_size=8B, tensor dims/strides, tile dims).  Groups 2/3 unused (<=2D).
__device__ __forceinline__ void tdm_load_rows(unsigned ldsByte, const __bf16* gsrc,
                                              unsigned rowUnits, unsigned nrows,
                                              unsigned long long strideUnits) {
  unsigned long long ga = (unsigned long long)(size_t)gsrc;
  v4u g0;
  g0[0] = 1u;                                              // count=1 (valid user D#)
  g0[1] = ldsByte;                                         // lds_addr
  g0[2] = (unsigned)(ga & 0xFFFFFFFFull);                  // global_addr[31:0]
  g0[3] = (unsigned)((ga >> 32) & 0x01FFFFFFull) | (2u << 30);  // global_addr[56:32], type=2
  unsigned td0 = (unsigned)strideUnits;                    // tensor_dim0 = full line
  unsigned td1 = nrows;                                    // tensor_dim1 = rows
  v8i g1 = {};
  g1[0] = (int)(3u << 16);                                 // data_size = 3 (8 bytes)
  g1[1] = (int)((td0 & 0xFFFFu) << 16);                    // tensor_dim0[15:0]
  g1[2] = (int)(((td0 >> 16) & 0xFFFFu) | ((td1 & 0xFFFFu) << 16));
  g1[3] = (int)(((td1 >> 16) & 0xFFFFu) | ((rowUnits & 0xFFFFu) << 16)); // tile_dim0
  g1[4] = (int)(nrows & 0xFFFFu);                          // tile_dim1
  g1[5] = (int)(strideUnits & 0xFFFFFFFFull);              // tensor_dim0_stride lo
  g1[6] = (int)((strideUnits >> 32) & 0xFFFFull);          // stride hi
  g1[7] = 0;
  v4i z4 = {};
#if defined(__clang_major__) && __clang_major__ >= 23
  v8i z8 = {};
  __builtin_amdgcn_tensor_load_to_lds(g0, g1, z4, z4, z8, 0);
#else
  __builtin_amdgcn_tensor_load_to_lds(g0, g1, z4, z4, 0);
#endif
}
#endif

// ---------------------------------------------------------------------------
// Pack OIHW f32 weights into WMMA A-fragment tiles (bf16), tap-major K:
// k = t*Cin + c.  A lane layout used here (contiguous K): lane L holds M=L%16,
// element e -> K_local = e + (L>=16 ? 16 : 0).
// packed[((mt*Ksteps+ks)*32 + lane)*16 + e]
// ---------------------------------------------------------------------------
__global__ void pack_w_kernel(const float* __restrict__ w, __bf16* __restrict__ packed,
                              int Cout, int Cin) {
  int Ksteps = (Cin * 9) >> 5;
  size_t total = (size_t)(Cout >> 4) * Ksteps * 512;
  size_t i = (size_t)blockIdx.x * blockDim.x + threadIdx.x;
  if (i >= total) return;
  int e = (int)(i & 15);
  int lane = (int)((i >> 4) & 31);
  size_t rest = i >> 9;
  int ks = (int)(rest % (size_t)Ksteps);
  int mt = (int)(rest / (size_t)Ksteps);
  int M = mt * 16 + (lane & 15);
  int kl = e + (lane >= 16 ? 16 : 0);
  int k = ks * 32 + kl;
  int t = k / Cin;
  int c = k - t * Cin;
  packed[i] = f2bf(w[((size_t)M * Cin + c) * 9 + t]);
}

// ---------------------------------------------------------------------------
// NCHW f32 -> NHWC bf16, LDS-tiled transpose. Grid: (HW/64, Cin/32, Nimg).
// ---------------------------------------------------------------------------
__global__ __launch_bounds__(256) void nchw2nhwc_kernel(const float* __restrict__ in,
                                                        __bf16* __restrict__ out, int Cin) {
  __shared__ float tile[32][65];
  const int p0 = blockIdx.x * 64;
  const int c0 = blockIdx.y * 32;
  const int img = blockIdx.z;
  const float* ip = in + ((size_t)img * Cin + c0) * HW_ + p0;
  const int tp = threadIdx.x & 63;
  const int tc = threadIdx.x >> 6;
#pragma unroll
  for (int r = 0; r < 32; r += 4)
    tile[tc + r][tp] = ip[(size_t)(tc + r) * HW_ + tp];
  __syncthreads();
  const int sc = threadIdx.x & 31;
  const int sp = threadIdx.x >> 5;
  __bf16* op = out + ((size_t)img * HW_ + p0) * Cin + c0;
#pragma unroll
  for (int r = 0; r < 64; r += 8)
    op[(size_t)(sp + r) * Cin + sc] = f2bf(tile[sc][sp + r]);
}

// ---------------------------------------------------------------------------
// Implicit-GEMM 3x3 SAME conv.  NHWC bf16 input, packed bf16 A-tiles staged to
// LDS by the Tensor Data Mover (double-buffered per 3x3 tap, s_wait_tensorcnt
// + workgroup barrier), f32 WMMA accumulate, NCHW f32 output.
// Grid: (32, Cout/64, Nimg). Block 256 = 8 waves.
// Wave: two 16-pixel columns x four 16-row M-tiles = 8 WMMAs per 32-K step.
// ---------------------------------------------------------------------------
__global__ __launch_bounds__(256) void conv3x3_wmma_kernel(
    const __bf16* __restrict__ inNHWC, const __bf16* __restrict__ wpack,
    const float* __restrict__ bias, const float* __restrict__ bnG,
    const float* __restrict__ bnB, const float* __restrict__ mulMask,
    float* __restrict__ out, int Cin, int Cout, int doRelu)
{
  extern __shared__ __bf16 aLDS[];                   // 2 x (4 tiles x csteps x 512)
  const int csteps = Cin >> 5;                       // K-steps per tap
  const int tapElems = 4 * csteps * 512;             // bf16 per LDS buffer
  const int lane = threadIdx.x & 31;
  const int wave = threadIdx.x >> 5;
  const int mq = blockIdx.y;
  const int img = blockIdx.z;
  const int p0 = (blockIdx.x * 8 + wave) * 32;       // 32 pixels per wave
  const int n0 = p0 + (lane & 15);
  const int n1 = n0 + 16;
  const int h  = p0 >> 7;                            // both halves share the row
  const int w0 = n0 & (WW_ - 1), w1 = n1 & (WW_ - 1);
  const __bf16* __restrict__ inimg = inNHWC + (size_t)img * HW_ * Cin;
  const int kb = (lane >= 16) ? 16 : 0;              // B frag: lanes 16-31 -> K 16..31
  const size_t mstride = (size_t)(9 * csteps) * 512; // packed elems per M-tile
  const __bf16* wqbase = wpack + (size_t)mq * 4 * mstride;

#if HAVE_TDM
  if (threadIdx.x == 0)
    tdm_load_rows(0u, wqbase, (unsigned)(csteps * 128), 4u,
                  (unsigned long long)(mstride >> 2));
  __builtin_amdgcn_s_wait_tensorcnt(0);
  __syncthreads();
#endif

  v8f acc[4][2];
#pragma unroll
  for (int mt = 0; mt < 4; ++mt) {
    acc[mt][0] = (v8f){};
    acc[mt][1] = (v8f){};
  }

#pragma unroll
  for (int t = 0; t < 9; ++t) {
#if HAVE_TDM
    if (t < 8 && threadIdx.x == 0)                   // prefetch next tap via TDM
      tdm_load_rows((unsigned)(((t + 1) & 1) * tapElems * 2),
                    wqbase + (size_t)(t + 1) * csteps * 512,
                    (unsigned)(csteps * 128), 4u,
                    (unsigned long long)(mstride >> 2));
#endif
    const int dh = t / 3, dw = t - (t / 3) * 3;
    const int hh = h + dh - 1;
    const int ww0 = w0 + dw - 1, ww1 = w1 + dw - 1;
    const bool okh = (hh >= 0) & (hh < HH_);
    const bool ok0 = okh & (ww0 >= 0) & (ww0 < WW_);
    const bool ok1 = okh & (ww1 >= 0) & (ww1 < WW_);
    const __bf16* b0p = inimg + ((size_t)(hh * WW_ + ww0)) * Cin + kb;
    const __bf16* b1p = inimg + ((size_t)(hh * WW_ + ww1)) * Cin + kb;
    for (int cs = 0; cs < csteps; ++cs) {
      v16bf b0v = {}; if (ok0) b0v = *(const v16bf*)(b0p + (size_t)cs * 32);
      v16bf b1v = {}; if (ok1) b1v = *(const v16bf*)(b1p + (size_t)cs * 32);
#if HAVE_TDM
      const __bf16* ap = aLDS + (size_t)(t & 1) * tapElems + (size_t)cs * 512 + lane * 16;
      const size_t tstride = (size_t)csteps * 512;   // M-tile stride inside buffer
#else
      const __bf16* ap = wqbase + ((size_t)t * csteps + cs) * 512 + lane * 16;
      const size_t tstride = mstride;
#endif
      v16bf av[4];
      av[0] = *(const v16bf*)(ap);
      av[1] = *(const v16bf*)(ap + tstride);
      av[2] = *(const v16bf*)(ap + 2 * tstride);
      av[3] = *(const v16bf*)(ap + 3 * tstride);
#pragma unroll
      for (int mt = 0; mt < 4; ++mt) {
        acc[mt][0] = __builtin_amdgcn_wmma_f32_16x16x32_bf16(false, av[mt], false, b0v,
                                                             (short)0, acc[mt][0], false, false);
        acc[mt][1] = __builtin_amdgcn_wmma_f32_16x16x32_bf16(false, av[mt], false, b1v,
                                                             (short)0, acc[mt][1], false, false);
      }
    }
#if HAVE_TDM
    if (t < 8) {
      __builtin_amdgcn_s_wait_tensorcnt(0);          // next tap slab landed
      __syncthreads();                               // all waves flip buffers
    }
#endif
  }

  const float RS = rsqrtf(1.0f + 1e-5f);
  const float mv0 = mulMask ? mulMask[(size_t)img * HW_ + n0] : 1.0f;
  const float mv1 = mulMask ? mulMask[(size_t)img * HW_ + n1] : 1.0f;
  float* __restrict__ outimg = out + (size_t)img * Cout * HW_;
  const int mrow = (lane >= 16) ? 8 : 0;             // C/D: VGPR r -> M=r(+8)
#pragma unroll
  for (int mt = 0; mt < 4; ++mt) {
#pragma unroll
    for (int r = 0; r < 8; ++r) {
      int M = mq * 64 + mt * 16 + mrow + r;
      float b = bias[M];
      float sA = bnG ? (bnG[M] * RS) : 1.0f;
      float sB = bnG ? bnB[M] : 0.0f;
      float y0 = (acc[mt][0][r] + b) * sA + sB;
      float y1 = (acc[mt][1][r] + b) * sA + sB;
      if (doRelu) { y0 = fmaxf(y0, 0.f); y1 = fmaxf(y1, 0.f); }
      outimg[(size_t)M * HW_ + n0] = y0 * mv0;
      outimg[(size_t)M * HW_ + n1] = y1 * mv1;
    }
  }
}

// ---------- glue kernels (bandwidth-trivial) ----------
__global__ void dot_rows_kernel(const float* __restrict__ q, const float* __restrict__ k,
                                float* __restrict__ s) {
  __shared__ float red[256];
  size_t base = (size_t)blockIdx.x * HW_;
  float acc = 0.f;
  for (int i = threadIdx.x; i < HW_; i += 256) acc += q[base + i] * k[base + i];
  red[threadIdx.x] = acc;
  __syncthreads();
  for (int off = 128; off > 0; off >>= 1) {
    if ((int)threadIdx.x < off) red[threadIdx.x] += red[threadIdx.x + off];
    __syncthreads();
  }
  if (threadIdx.x == 0) s[blockIdx.x] = red[0];
}

__global__ void softmax64_kernel(float* __restrict__ s) {
  __shared__ float buf[64];
  int g = blockIdx.x, c = threadIdx.x;
  float v = s[g * 64 + c];
  buf[c] = v; __syncthreads();
  float m = buf[0];
  for (int i = 1; i < 64; ++i) m = fmaxf(m, buf[i]);
  float e = __expf(v - m);
  __syncthreads();
  buf[c] = e; __syncthreads();
  float sum = 0.f;
  for (int i = 0; i < 64; ++i) sum += buf[i];
  s[g * 64 + c] = e / sum;
}

__global__ void ctx_mean_kernel(const float* __restrict__ attn, const float* __restrict__ v,
                                float* __restrict__ ctx, int nmaps, int baseMul, int mapStride,
                                size_t total) {
  size_t i = (size_t)blockIdx.x * blockDim.x + threadIdx.x;
  if (i >= total) return;
  int p = (int)(i & (HW_ - 1));
  int c = (int)((i >> 13) & 63);
  int g = (int)(i >> 19);
  float s = 0.f;
  for (int j = 0; j < nmaps; ++j) {
    int im = g * baseMul + j * mapStride;
    s += attn[im * 64 + c] * v[((size_t)im * 64 + c) * HW_ + p];
  }
  ctx[i] = s / (float)nmaps;
}

__global__ void mask_head_kernel(const float* __restrict__ m, const float* __restrict__ w2,
                                 const float* __restrict__ b2, const float* __restrict__ g2,
                                 const float* __restrict__ be2, float* __restrict__ mask) {
  int i = blockIdx.x * blockDim.x + threadIdx.x;
  if (i >= 4 * HW_) return;
  int b = i >> 13, p = i & (HW_ - 1);
  int h = p >> 7, x = p & (WW_ - 1);
  float s = 0.f;
  for (int c = 0; c < 64; ++c) {
    const float* mc = m + ((size_t)b * 64 + c) * HW_;
#pragma unroll
    for (int t = 0; t < 9; ++t) {
      int hh = h + t / 3 - 1, ww = x + (t - (t / 3) * 3) - 1;
      if (hh >= 0 && hh < HH_ && ww >= 0 && ww < WW_)
        s += w2[c * 9 + t] * mc[hh * WW_ + ww];
    }
  }
  s += b2[0];
  s = s * (g2[0] * rsqrtf(1.f + 1e-5f)) + be2[0];
  mask[i] = 1.f / (1.f + __expf(-s));
}

__global__ void add_kernel(const float* __restrict__ a, const float* __restrict__ b,
                           float* __restrict__ o, size_t total) {
  size_t i = (size_t)blockIdx.x * blockDim.x + threadIdx.x;
  if (i < total) o[i] = a[i] + b[i];
}

// ---------------------------------------------------------------------------
extern "C" void kernel_launch(void* const* d_in, const int* in_sizes, int n_in,
                              void* d_out, int out_size, void* d_ws, size_t ws_size,
                              hipStream_t stream) {
  (void)in_sizes; (void)n_in; (void)out_size; (void)ws_size;
  enum {
    IN_FEAT = 0, IN_CNT,
    SP_WQ, SP_BQ, SP_WK, SP_BK, SP_WV, SP_BV,
    SP_WO1, SP_BO1, SP_G1, SP_BE1,
    SP_WO2, SP_BO2, SP_G2, SP_BE2,
    TP_WQ, TP_BQ, TP_WK, TP_BK, TP_WV, TP_BV,
    TP_WO1, TP_BO1, TP_G1, TP_BE1,
    TP_WO2, TP_BO2, TP_G2, TP_BE2,
    TP_WG, TP_BG,
    MK_W1, MK_B1, MK_G1, MK_BE1,
    MK_W2, MK_B2, MK_G2, MK_BE2,
    OUT_W, OUT_B, OUT_G, OUT_BE
  };
  auto P = [&](int i) { return (const float*)d_in[i]; };
  const float* feat = P(IN_FEAT);

  // ---- workspace arena (floats) ----
  constexpr size_t IMG64  = 64ull * HW_;
  constexpr size_t IMG256 = 256ull * HW_;
  constexpr size_t OFF_Q   = 0;
  constexpr size_t OFF_K   = OFF_Q   + 48 * IMG64;
  constexpr size_t OFF_V   = OFF_K   + 48 * IMG64;
  constexpr size_t OFF_CTX = OFF_V   + 48 * IMG64;
  constexpr size_t OFF_O1  = OFF_CTX + 12 * IMG64;
  constexpr size_t OFF_SP  = OFF_O1  + 12 * IMG256;
  constexpr size_t OFF_SC  = OFF_SP  + 12 * IMG256;
  constexpr size_t OFF_WP  = OFF_SC  + 4096;            // packed weights (bf16)
  constexpr size_t WP_FLOATS = (9 * 147456ull + 4 * 589824ull + 1) / 2;
  constexpr size_t OFF_NB  = OFF_WP + WP_FLOATS;        // NHWC bf16 staging
  float* Wf = (float*)d_ws;
  __bf16* WPbase = (__bf16*)(Wf + OFF_WP);
  __bf16* NB     = (__bf16*)(Wf + OFF_NB);              // up to 48*8192*256 bf16

  // ---- pack conv weights into WMMA A-tile layout (bf16, tap-major K) ----
  size_t wpo = 0;
  auto pack = [&](int idx, int Cout, int Cin) -> __bf16* {
    __bf16* dst = WPbase + wpo;
    size_t total = (size_t)(Cout / 16) * (Cin * 9 / 32) * 512;
    wpo += total;
    pack_w_kernel<<<dim3((unsigned)((total + 255) / 256)), dim3(256), 0, stream>>>(
        P(idx), dst, Cout, Cin);
    return dst;
  };
  __bf16 *wq  = pack(SP_WQ, 64, 256),  *wk  = pack(SP_WK, 64, 256),  *wv  = pack(SP_WV, 64, 256);
  __bf16 *wo1 = pack(SP_WO1, 256, 64), *wo2 = pack(SP_WO2, 256, 256);
  __bf16 *twq = pack(TP_WQ, 64, 256),  *twk = pack(TP_WK, 64, 256),  *twv = pack(TP_WV, 64, 256);
  __bf16 *two1 = pack(TP_WO1, 256, 64), *two2 = pack(TP_WO2, 256, 256);
  __bf16 *twg = pack(TP_WG, 256, 256), *wm1 = pack(MK_W1, 64, 256), *wout = pack(OUT_W, 256, 256);

  auto cvt = [&](const float* src, int nimg, int Cin) {   // NCHW f32 -> NHWC bf16
    nchw2nhwc_kernel<<<dim3(HW_ / 64, (unsigned)(Cin / 32), (unsigned)nimg),
                       dim3(256), 0, stream>>>(src, NB, Cin);
  };
  auto conv = [&](const __bf16* wp, int bidx, const float* g, const float* be,
                  const float* mm, float* out, int Cin, int Cout, int relu, int nimg) {
    unsigned shmem = 2u * 4u * (unsigned)(Cin / 32) * 512u * 2u;   // double-buffered A
    conv3x3_wmma_kernel<<<dim3(32, (unsigned)(Cout / 64), (unsigned)nimg),
                          dim3(256), shmem, stream>>>(NB, wp, P(bidx), g, be, mm, out,
                                                      Cin, Cout, relu);
  };

  float* bq  = Wf + OFF_Q;  float* bk = Wf + OFF_K;  float* bv = Wf + OFF_V;
  float* ctx = Wf + OFF_CTX; float* o1 = Wf + OFF_O1; float* sp = Wf + OFF_SP;
  float* sc  = Wf + OFF_SC;
  float* outMask  = (float*)d_out;           // [4, 64, 128]
  float* outFused = outMask + 4 * HW_;       // [4, 256, 64, 128]

  // ---------------- spatial fusion ----------------
  cvt(feat, 48, 256);
  conv(wq, SP_BQ, nullptr, nullptr, nullptr, bq, 256, 64, 0, 48);
  conv(wk, SP_BK, nullptr, nullptr, nullptr, bk, 256, 64, 0, 48);
  conv(wv, SP_BV, nullptr, nullptr, nullptr, bv, 256, 64, 0, 48);
  dot_rows_kernel<<<dim3(48 * 64), dim3(256), 0, stream>>>(bq, bk, sc);
  softmax64_kernel<<<dim3(48), dim3(64), 0, stream>>>(sc);
  {
    size_t total = 12ull * 64 * HW_;
    ctx_mean_kernel<<<dim3((unsigned)((total + 255) / 256)), dim3(256), 0, stream>>>(
        sc, bv, ctx, 4, 4, 1, total);   // im = g*4 + j
  }
  cvt(ctx, 12, 64);
  conv(wo1, SP_BO1, P(SP_G1), P(SP_BE1), nullptr, o1, 64, 256, 1, 12);
  cvt(o1, 12, 256);
  conv(wo2, SP_BO2, P(SP_G2), P(SP_BE2), nullptr, sp, 256, 256, 1, 12);

  // ---------------- temporal fusion ----------------
  cvt(sp, 12, 256);
  conv(twq, TP_BQ, nullptr, nullptr, nullptr, bq, 256, 64, 0, 12);
  conv(twk, TP_BK, nullptr, nullptr, nullptr, bk, 256, 64, 0, 12);
  conv(twv, TP_BV, nullptr, nullptr, nullptr, bv, 256, 64, 0, 12);
  dot_rows_kernel<<<dim3(12 * 64), dim3(256), 0, stream>>>(bq, bk, sc);
  softmax64_kernel<<<dim3(12), dim3(64), 0, stream>>>(sc);
  {
    size_t total = 4ull * 64 * HW_;
    ctx_mean_kernel<<<dim3((unsigned)((total + 255) / 256)), dim3(256), 0, stream>>>(
        sc, bv, ctx, 3, 1, 4, total);   // im = b + t*4
  }
  cvt(ctx, 4, 64);
  conv(two1, TP_BO1, P(TP_G1), P(TP_BE1), nullptr, o1, 64, 256, 1, 4);
  float* tpo2 = bq;   // dead q/k/v regions reused
  float* diff = bk;
  float* tout = bv;
  cvt(o1, 4, 256);
  conv(two2, TP_BO2, P(TP_G2), P(TP_BE2), nullptr, tpo2, 256, 256, 1, 4);
  cvt(tpo2, 4, 256);
  conv(twg, TP_BG, nullptr, nullptr, nullptr, diff, 256, 256, 0, 4);

  // ---------------- mask head ----------------
  float* mbuf = ctx;  // ctx_tp dead
  cvt(diff, 4, 256);
  conv(wm1, MK_B1, P(MK_G1), P(MK_BE1), nullptr, mbuf, 256, 64, 1, 4);
  mask_head_kernel<<<dim3((4 * HW_) / 256), dim3(256), 0, stream>>>(
      mbuf, P(MK_W2), P(MK_B2), P(MK_G2), P(MK_BE2), outMask);

  // ---------------- residual + fused output ----------------
  {
    size_t total = 4ull * 256 * HW_;   // temporal_out = diff + spatial[t=0]
    add_kernel<<<dim3((unsigned)((total + 255) / 256)), dim3(256), 0, stream>>>(
        diff, sp, tout, total);
  }
  cvt(tout, 4, 256);
  conv(wout, OUT_B, P(OUT_G), P(OUT_BE), outMask, outFused, 256, 256, 1, 4);
}
